// MultiHeadedAttention_64854006169846
// MI455X (gfx1250) — matrix operand order repaired and checked by reading, not compiled, not run on hardware
//
#include <hip/hip_runtime.h>
#include <hip/hip_bf16.h>
#include <stdint.h>

#define BB   2
#define SS   2048
#define DD   1024
#define HH   16
#define DHH  64
#define NEGV (-1e18f)

typedef __bf16 bf16_t;
typedef __attribute__((ext_vector_type(16))) __bf16 v16bf;
typedef __attribute__((ext_vector_type(4)))  __bf16 v4bf;
typedef __attribute__((ext_vector_type(8)))  float  v8f;

// GCC-style vector typedefs matching the async-LDS builtin's parameter types
typedef int v4i_vs __attribute__((vector_size(16)));
typedef __attribute__((address_space(1))) v4i_vs global_v4i;
typedef __attribute__((address_space(3))) v4i_vs lds_v4i;

#if __has_builtin(__builtin_amdgcn_global_load_async_to_lds_b128)
#define HAVE_ASYNC_LDS 1
#else
#define HAVE_ASYNC_LDS 0
#endif

// A-operand (16x32 bf16): lane(halfbit, m): elems 0-15 = A[m][half*16+i]
__device__ __forceinline__ v16bf ldA16(const bf16_t* p0, const bf16_t* p1) {
  v16bf r;
#pragma unroll
  for (int i = 0; i < 8; i++) { r[i] = p0[i]; r[i + 8] = p1[i]; }
  return r;
}
// B-operand (32x16 bf16): lane(halfbit, n): elems 0-15 = B[half*16+i][n] = Bt[n][half*16+i] contiguous
__device__ __forceinline__ v16bf ldB16(const bf16_t* p) {
  v16bf r;
#pragma unroll
  for (int i = 0; i < 16; i++) r[i] = p[i];
  return r;
}
__device__ __forceinline__ v8f wmma_bf16(v16bf a, v16bf b, v8f c) {
  return __builtin_amdgcn_wmma_f32_16x16x32_bf16(false, a, false, b, (short)0, c, false, false);
}

// ---------------------------------------------------------------------------
// Kernel 1: QKV projections. z=0: Q (scaled 0.125), z=1: K, z=2: V (transposed)
// ---------------------------------------------------------------------------
__global__ __launch_bounds__(128)
void qkv_proj_kernel(const float* __restrict__ q_in, const float* __restrict__ k_in,
                     const float* __restrict__ v_in,
                     const float* __restrict__ Wq, const float* __restrict__ bq,
                     const float* __restrict__ Wk, const float* __restrict__ bk,
                     const float* __restrict__ Wv, const float* __restrict__ bv,
                     bf16_t* __restrict__ qh, bf16_t* __restrict__ kh,
                     bf16_t* __restrict__ vth) {
  const int z = blockIdx.z;
  const float* X    = (z == 0) ? q_in : (z == 1) ? k_in : v_in;
  const float* W    = (z == 0) ? Wq   : (z == 1) ? Wk   : Wv;
  const float* bias = (z == 0) ? bq   : (z == 1) ? bk   : bv;

  const int m0 = blockIdx.x * 64;
  const int n0 = blockIdx.y * 64;
  const int tid = threadIdx.x;
  const int wave = tid >> 5;
  const int lane = tid & 31;
  const int hb = lane >> 4;   // wave half
  const int hl = lane & 15;   // lane within half

  __shared__ bf16_t Xl[64 * 32];
  __shared__ bf16_t Wl[64 * 32];

  v8f c[4] = {};

  for (int k0 = 0; k0 < DD; k0 += 32) {
    __syncthreads();
#pragma unroll
    for (int i = 0; i < 4; i++) {
      int idx = tid + i * 128;               // 0..511 float4 slots
      int r = idx >> 3, c4 = (idx & 7) * 4;  // 8 float4 per 32-col row
      float4 xv = *reinterpret_cast<const float4*>(&X[(size_t)(m0 + r) * DD + k0 + c4]);
      float4 wv = *reinterpret_cast<const float4*>(&W[(size_t)(n0 + r) * DD + k0 + c4]);
      v4bf xb = {(bf16_t)xv.x, (bf16_t)xv.y, (bf16_t)xv.z, (bf16_t)xv.w};
      v4bf wb = {(bf16_t)wv.x, (bf16_t)wv.y, (bf16_t)wv.z, (bf16_t)wv.w};
      *reinterpret_cast<v4bf*>(&Xl[r * 32 + c4]) = xb;
      *reinterpret_cast<v4bf*>(&Wl[r * 32 + c4]) = wb;
    }
    __syncthreads();

    const bf16_t* arow = &Xl[(wave * 16 + hl) * 32];
    v16bf a = ldA16(arow + hb * 16, arow + 8 + hb * 16);
#pragma unroll
    for (int nt = 0; nt < 4; nt++) {
      v16bf bm = ldB16(&Wl[(nt * 16 + hl) * 32 + hb * 16]);
      c[nt] = wmma_bf16(a, bm, c[nt]);
    }
  }

#pragma unroll
  for (int nt = 0; nt < 4; nt++) {
    int n = n0 + nt * 16 + hl;
    float bval = bias[n];
    int h = n >> 6, dh = n & 63;
#pragma unroll
    for (int j = 0; j < 8; j++) {
      int M = m0 + wave * 16 + j + 8 * hb;
      int bidx = M >> 11, s = M & (SS - 1);
      float val = c[nt][j] + bval;
      if (z == 0)
        qh[((((size_t)bidx * HH + h) * SS + s) << 6) + dh] = (bf16_t)(val * 0.125f);
      else if (z == 1)
        kh[((((size_t)bidx * HH + h) * SS + s) << 6) + dh] = (bf16_t)val;
      else
        vth[(((size_t)bidx * HH + h) * DHH + dh) * SS + s] = (bf16_t)val;
    }
  }
}

// ---------------------------------------------------------------------------
// Kernel 2: flash attention, double-buffered K/V tiles via async LDS loads.
// ---------------------------------------------------------------------------
__device__ __forceinline__ void issue_kv_tile(bf16_t* Kbuf, bf16_t* Vbuf,
                                              const bf16_t* __restrict__ kh,
                                              const bf16_t* __restrict__ vth,
                                              size_t kBase, size_t vBase, int k0, int tid) {
#pragma unroll
  for (int i = 0; i < 4; i++) {
    int idx = tid + i * 128;           // 0..511 16-byte slots
    int row = idx >> 3, u = idx & 7;   // 8 x 16B per 64-elem bf16 row
    const bf16_t* gk = kh + kBase + (size_t)(k0 + row) * DHH + u * 8;
    const bf16_t* gv = vth + vBase + (size_t)row * SS + k0 + u * 8;
#if HAVE_ASYNC_LDS
    __builtin_amdgcn_global_load_async_to_lds_b128(
        (global_v4i*)gk, (lds_v4i*)(Kbuf + (size_t)idx * 8), 0, 0);
    __builtin_amdgcn_global_load_async_to_lds_b128(
        (global_v4i*)gv, (lds_v4i*)(Vbuf + (size_t)idx * 8), 0, 0);
#else
    reinterpret_cast<uint4*>(Kbuf)[idx] = *reinterpret_cast<const uint4*>(gk);
    reinterpret_cast<uint4*>(Vbuf)[idx] = *reinterpret_cast<const uint4*>(gv);
#endif
  }
}

__device__ __forceinline__ void wait_async_lds() {
#if HAVE_ASYNC_LDS
#if __has_builtin(__builtin_amdgcn_s_wait_asynccnt)
  __builtin_amdgcn_s_wait_asynccnt(0);
#else
  asm volatile("s_wait_asynccnt 0x0" ::: "memory");
#endif
#endif
}

__global__ __launch_bounds__(128)
void flash_attn_kernel(const bf16_t* __restrict__ qh, const bf16_t* __restrict__ kh,
                       const bf16_t* __restrict__ vth, const uint8_t* __restrict__ mask,
                       bf16_t* __restrict__ ctx, float* __restrict__ topScores) {
  const int b = blockIdx.z, h = blockIdx.y;
  const int q0 = blockIdx.x * 64;
  const int tid = threadIdx.x;
  const int wave = tid >> 5;
  const int lane = tid & 31;
  const int hb = lane >> 4;
  const int hl = lane & 15;
  const int qbase = q0 + wave * 16;

  __shared__ bf16_t Kl[2][64 * 64];    // [buf][key][dh]
  __shared__ bf16_t Vl[2][64 * 64];    // [buf][dh][key]  (V transposed)
  __shared__ bf16_t Pl[4][16 * 64];    // per-wave P staging

  const size_t kBase = ((size_t)b * HH + h) * (size_t)SS * DHH;
  const size_t vBase = ((size_t)b * HH + h) * (size_t)DHH * SS;
  const bf16_t* qrow = qh + kBase + (size_t)(qbase + hl) * DHH;
  v16bf qa0 = ldA16(qrow + hb * 16,      qrow + 8 + hb * 16);
  v16bf qa1 = ldA16(qrow + 32 + hb * 16, qrow + 40 + hb * 16);

  v8f acc[4] = {};
  float run_m[8], run_l[8];
#pragma unroll
  for (int j = 0; j < 8; j++) { run_m[j] = -__builtin_inff(); run_l[j] = 0.0f; }

  const size_t maskBase = (size_t)b * SS * SS;

  // prefetch first tile
  issue_kv_tile(Kl[0], Vl[0], kh, vth, kBase, vBase, 0, tid);

  const int NB = SS / 64;
  for (int kb = 0; kb < NB; kb++) {
    const int k0 = kb * 64;
    const int cur = kb & 1;

    wait_async_lds();     // own async writes to LDS complete
    __syncthreads();      // tile published; previous iteration's reads done
    if (kb + 1 < NB)
      issue_kv_tile(Kl[cur ^ 1], Vl[cur ^ 1], kh, vth, kBase, vBase, k0 + 64, tid);

    // scores = Q * K^T  (scale folded into Q)
    float sc[4][8];
#pragma unroll
    for (int nt = 0; nt < 4; nt++) {
      const bf16_t* kr = &Kl[cur][(nt * 16 + hl) * 64];
      v16bf b0 = ldB16(kr + hb * 16);
      v16bf b1 = ldB16(kr + 32 + hb * 16);
      v8f cc = {};
      cc = wmma_bf16(qa0, b0, cc);
      cc = wmma_bf16(qa1, b1, cc);
      int key = k0 + nt * 16 + hl;
#pragma unroll
      for (int j = 0; j < 8; j++) {
        int q = qbase + j + 8 * hb;
        uint8_t mv = mask[maskBase + (size_t)q * SS + key];
        float s = mv ? NEGV : cc[j];
        sc[nt][j] = s;
        if (h == 0) topScores[((size_t)b * SS + q) * SS + key] = s;
      }
    }

    // online softmax (row = 16 lanes of one wave half, reduce via shfl_xor)
#pragma unroll
    for (int j = 0; j < 8; j++) {
      float mb = fmaxf(fmaxf(sc[0][j], sc[1][j]), fmaxf(sc[2][j], sc[3][j]));
#pragma unroll
      for (int mks = 1; mks < 16; mks <<= 1) mb = fmaxf(mb, __shfl_xor(mb, mks, 32));
      float nm = fmaxf(run_m[j], mb);
      float alpha = __expf(run_m[j] - nm);
      run_m[j] = nm;
      float ls = 0.0f;
#pragma unroll
      for (int nt = 0; nt < 4; nt++) {
        float p = __expf(sc[nt][j] - nm);
        sc[nt][j] = p;
        ls += p;
      }
#pragma unroll
      for (int mks = 1; mks < 16; mks <<= 1) ls += __shfl_xor(ls, mks, 32);
      run_l[j] = run_l[j] * alpha + ls;
#pragma unroll
      for (int d = 0; d < 4; d++) acc[d][j] *= alpha;
    }

    // C-layout -> A-layout via per-wave LDS round trip (LDS in-order per wave)
#pragma unroll
    for (int nt = 0; nt < 4; nt++)
#pragma unroll
      for (int j = 0; j < 8; j++)
        Pl[wave][(j + 8 * hb) * 64 + nt * 16 + hl] = (bf16_t)sc[nt][j];

    const bf16_t* prow = &Pl[wave][hl * 64];
    v16bf pa0 = ldA16(prow + hb * 16,      prow + 8 + hb * 16);
    v16bf pa1 = ldA16(prow + 32 + hb * 16, prow + 40 + hb * 16);
#pragma unroll
    for (int d = 0; d < 4; d++) {
      const bf16_t* vr = &Vl[cur][(d * 16 + hl) * 64];
      v16bf vb0 = ldB16(vr + hb * 16);
      v16bf vb1 = ldB16(vr + 32 + hb * 16);
      acc[d] = wmma_bf16(pa0, vb0, acc[d]);
      acc[d] = wmma_bf16(pa1, vb1, acc[d]);
    }
  }

  // finalize: ctx in (B, S, H*DH) order, bf16 for the out-projection GEMM
#pragma unroll
  for (int j = 0; j < 8; j++) {
    float inv = 1.0f / run_l[j];
    int q = qbase + j + 8 * hb;
#pragma unroll
    for (int d = 0; d < 4; d++) {
      int dh = d * 16 + hl;
      ctx[((size_t)b * SS + q) * DD + h * DHH + dh] = (bf16_t)(acc[d][j] * inv);
    }
  }
}

// ---------------------------------------------------------------------------
// Kernel 3: exact row softmax over the raw head-0 scores (in place in d_out)
// ---------------------------------------------------------------------------
__global__ __launch_bounds__(256)
void softmax_rows_kernel(float* __restrict__ top) {
  float* p = top + (size_t)blockIdx.x * SS;
  const int tid = threadIdx.x;
  __shared__ float red[256];

  float mx = -__builtin_inff();
  for (int i = tid; i < SS; i += 256) mx = fmaxf(mx, p[i]);
  red[tid] = mx; __syncthreads();
  for (int s = 128; s > 0; s >>= 1) { if (tid < s) red[tid] = fmaxf(red[tid], red[tid + s]); __syncthreads(); }
  mx = red[0]; __syncthreads();

  float sum = 0.0f;
  for (int i = tid; i < SS; i += 256) { float e = __expf(p[i] - mx); p[i] = e; sum += e; }
  red[tid] = sum; __syncthreads();
  for (int s = 128; s > 0; s >>= 1) { if (tid < s) red[tid] += red[tid + s]; __syncthreads(); }
  float inv = 1.0f / red[0];
  for (int i = tid; i < SS; i += 256) p[i] *= inv;
}

// ---------------------------------------------------------------------------
// Kernel 4: out = ctx @ Wo^T + bo   (bf16 WMMA, f32 out)
// ---------------------------------------------------------------------------
__global__ __launch_bounds__(128)
void out_proj_kernel(const bf16_t* __restrict__ ctx, const float* __restrict__ Wo,
                     const float* __restrict__ bo, float* __restrict__ out) {
  const int m0 = blockIdx.x * 64;
  const int n0 = blockIdx.y * 64;
  const int tid = threadIdx.x;
  const int wave = tid >> 5;
  const int lane = tid & 31;
  const int hb = lane >> 4;
  const int hl = lane & 15;

  __shared__ bf16_t Xl[64 * 32];
  __shared__ bf16_t Wl[64 * 32];

  v8f c[4] = {};

  for (int k0 = 0; k0 < DD; k0 += 32) {
    __syncthreads();
#pragma unroll
    for (int i = 0; i < 2; i++) {
      int idx = tid + i * 128;           // 256 uint4 = 64 rows * 32 bf16
      int r = idx >> 2, u = idx & 3;
      reinterpret_cast<uint4*>(Xl)[idx] =
          *reinterpret_cast<const uint4*>(ctx + (size_t)(m0 + r) * DD + k0 + u * 8);
    }
#pragma unroll
    for (int i = 0; i < 4; i++) {
      int idx = tid + i * 128;
      int r = idx >> 3, c4 = (idx & 7) * 4;
      float4 wv = *reinterpret_cast<const float4*>(&Wo[(size_t)(n0 + r) * DD + k0 + c4]);
      v4bf wb = {(bf16_t)wv.x, (bf16_t)wv.y, (bf16_t)wv.z, (bf16_t)wv.w};
      *reinterpret_cast<v4bf*>(&Wl[r * 32 + c4]) = wb;
    }
    __syncthreads();

    const bf16_t* arow = &Xl[(wave * 16 + hl) * 32];
    v16bf a = ldA16(arow + hb * 16, arow + 8 + hb * 16);
#pragma unroll
    for (int nt = 0; nt < 4; nt++) {
      v16bf bm = ldB16(&Wl[(nt * 16 + hl) * 32 + hb * 16]);
      c[nt] = wmma_bf16(a, bm, c[nt]);
    }
  }

#pragma unroll
  for (int nt = 0; nt < 4; nt++) {
    int n = n0 + nt * 16 + hl;
    float bval = bo[n];
#pragma unroll
    for (int j = 0; j < 8; j++) {
      int M = m0 + wave * 16 + j + 8 * hb;
      out[(size_t)M * DD + n] = c[nt][j] + bval;
    }
  }
}

// ---------------------------------------------------------------------------
extern "C" void kernel_launch(void* const* d_in, const int* in_sizes, int n_in,
                              void* d_out, int out_size, void* d_ws, size_t ws_size,
                              hipStream_t stream) {
  const float*   key   = (const float*)d_in[0];
  const float*   value = (const float*)d_in[1];
  const float*   query = (const float*)d_in[2];
  const uint8_t* mask  = (const uint8_t*)d_in[3];
  const float*   Wk    = (const float*)d_in[4];
  const float*   bk    = (const float*)d_in[5];
  const float*   Wv    = (const float*)d_in[6];
  const float*   bv    = (const float*)d_in[7];
  const float*   Wq    = (const float*)d_in[8];
  const float*   bq    = (const float*)d_in[9];
  const float*   Wo    = (const float*)d_in[10];
  const float*   bo    = (const float*)d_in[11];

  float* out = (float*)d_out;                        // (B,S,D)
  float* top = out + (size_t)BB * SS * DD;           // (B,S,S) head-0 attn

  const size_t elems = (size_t)BB * HH * SS * DHH;   // 4M per buffer
  bf16_t* qh  = (bf16_t*)d_ws;
  bf16_t* kh  = qh + elems;
  bf16_t* vth = kh + elems;
  bf16_t* ctx = vth + elems;                         // (B,S,D) bf16

  qkv_proj_kernel<<<dim3(64, 16, 3), 128, 0, stream>>>(query, key, value, Wq, bq, Wk, bk,
                                                       Wv, bv, qh, kh, vth);
  flash_attn_kernel<<<dim3(SS / 64, HH, BB), 128, 0, stream>>>(qh, kh, vth, mask, ctx, top);
  softmax_rows_kernel<<<BB * SS, 256, 0, stream>>>(top);
  out_proj_kernel<<<dim3(64, 16, 1), 128, 0, stream>>>(ctx, Wo, bo, out);
}
